// SparseBlock_65223373357584
// MI455X (gfx1250) — matrix operand order repaired and checked by reading, not compiled, not run on hardware
//
#include <hip/hip_runtime.h>

typedef __bf16 bf16;
typedef __attribute__((ext_vector_type(16))) __bf16 v16bf;
typedef __attribute__((ext_vector_type(8)))  __bf16 v8bf;
typedef __attribute__((ext_vector_type(8)))  float   v8f;
typedef __attribute__((ext_vector_type(4)))  unsigned int v4u;

#define CIN   128
#define COUT  128
#define KTAPS 9
#define LDSW_STRIDE 136   // 128 + 8 bf16 pad -> 272B rows, avoids LDS bank conflicts
#define EPS   1e-4f
#define LEAK  0.333f

// ---------- prep 1: feats f32 -> bf16, plus zero row at index N ----------
__global__ __launch_bounds__(256) void cvt_feats_kernel(const float* __restrict__ src,
                                                        bf16* __restrict__ dst,
                                                        long n_valid, long n_total)
{
    long base = ((long)blockIdx.x * blockDim.x + threadIdx.x) * 8;
    if (base >= n_total) return;
    v8bf o;
    if (base < n_valid) {
        float4 f0 = *(const float4*)(src + base);
        float4 f1 = *(const float4*)(src + base + 4);
        o[0]=(bf16)f0.x; o[1]=(bf16)f0.y; o[2]=(bf16)f0.z; o[3]=(bf16)f0.w;
        o[4]=(bf16)f1.x; o[5]=(bf16)f1.y; o[6]=(bf16)f1.z; o[7]=(bf16)f1.w;
    } else {
        #pragma unroll
        for (int i = 0; i < 8; ++i) o[i] = (bf16)0.0f;
    }
    *(v8bf*)(dst + base) = o;
}

// ---------- prep 2: W[k][cin][cout] f32 -> Wt[k][cout][cin] bf16 ----------
__global__ __launch_bounds__(256) void cvt_w_kernel(const float* __restrict__ W,
                                                    bf16* __restrict__ wt)
{
    int e = blockIdx.x * blockDim.x + threadIdx.x;
    if (e >= KTAPS * CIN * COUT) return;
    int k = e / (CIN * COUT);
    int r = e % (CIN * COUT);
    int n = r / CIN;     // cout
    int c = r % CIN;     // cin
    wt[e] = (bf16)W[k * CIN * COUT + c * COUT + n];
}

// A fragment loader: 16x32 bf16 tile, K-base kb, per ISA 7.12.2 VGPR layout
__device__ __forceinline__ v16bf load_a_frag(const bf16* __restrict__ arow, int kb, int hi)
{
    v8bf a0 = *(const v8bf*)(arow + kb + hi * 8);
    v8bf a1 = *(const v8bf*)(arow + kb + 16 + hi * 8);
    return __builtin_shufflevector(a0, a1, 0,1,2,3,4,5,6,7,8,9,10,11,12,13,14,15);
}

// ---------- main: gather + 9-tap GEMM via bf16 WMMA ----------
// Block = 256 threads (8 wave32), covers 256 output rows: each wave owns two
// 16-row M-tiles so every B fragment (LDS) feeds two WMMAs.
__global__ __launch_bounds__(256) void sparse_gemm_kernel(const bf16* __restrict__ fbf,
                                                          const bf16* __restrict__ wt,
                                                          const float* __restrict__ bias,
                                                          const int* __restrict__ neigh,
                                                          float* __restrict__ out,
                                                          long N)
{
    __shared__ bf16 ldsW[COUT * LDSW_STRIDE];   // 34816 B

    const int t    = threadIdx.x;
    const int lane = t & 31;
    const int wv   = t >> 5;        // wave 0..7
    const int m    = lane & 15;     // M-row / N-col within 16x16 tile
    const int hi   = lane >> 4;     // K-half select per WMMA VGPR layout
    const long blockRow = (long)blockIdx.x * 256;
    const long rowBase  = blockRow + wv * 32;
    const long gm0 = rowBase + m;          // M-tile 0 gather row
    const long gm1 = rowBase + 16 + m;     // M-tile 1 gather row
    const long gc0 = gm0 < N ? gm0 : N - 1;   // clamp (tail block): keep EXEC uniform
    const long gc1 = gm1 < N ? gm1 : N - 1;

    // preload all neighbor indices up front (kills per-tap load->use chain)
    int ridx0[KTAPS], ridx1[KTAPS];
    #pragma unroll
    for (int k = 0; k < KTAPS; ++k) {
        ridx0[k] = neigh[gc0 * KTAPS + k];
        ridx1[k] = neigh[gc1 * KTAPS + k];
    }

    // accumulators: 2 M-tiles x 8 N-tiles of 16x16 f32, init with bias[col]
    v8f accA[8], accB[8];
    #pragma unroll
    for (int nt = 0; nt < 8; ++nt) {
        float bv = bias[nt * 16 + m];
        accA[nt] = (v8f){bv, bv, bv, bv, bv, bv, bv, bv};
        accB[nt] = accA[nt];
    }

    for (int k = 0; k < KTAPS; ++k) {
        __syncthreads();
        {   // stage Wt[k] (32KB) into padded LDS
            const v4u* src = (const v4u*)(wt + (size_t)k * CIN * COUT);
            #pragma unroll
            for (int j = 0; j < 8; ++j) {
                int cid = t + j * 256;          // 16B chunk id, 2048 total
                int row = cid >> 4;
                int col = cid & 15;
                *((v4u*)(ldsW + row * LDSW_STRIDE) + col) = src[cid];
            }
        }
        __syncthreads();

        const bf16* arow0 = fbf + (size_t)ridx0[k] * CIN;   // sentinel N -> zero row
        const bf16* arow1 = fbf + (size_t)ridx1[k] * CIN;

        #pragma unroll
        for (int kb = 0; kb < CIN; kb += 32) {
            v16bf a0 = load_a_frag(arow0, kb, hi);
            v16bf a1 = load_a_frag(arow1, kb, hi);
            #pragma unroll
            for (int nt = 0; nt < 8; ++nt) {
                // B fragment: lane holds col n = nt*16+m, K = kb+hi*8..+7 and kb+16+hi*8..+7
                const bf16* bp = ldsW + (nt * 16 + m) * LDSW_STRIDE + kb + hi * 8;
                v8bf b0 = *(const v8bf*)bp;
                v8bf b1 = *(const v8bf*)(bp + 16);
                v16bf b = __builtin_shufflevector(b0, b1,
                            0,1,2,3,4,5,6,7,8,9,10,11,12,13,14,15);
                accA[nt] = __builtin_amdgcn_wmma_f32_16x16x32_bf16(
                               false, a0, false, b, (short)0, accA[nt], false, false);
                accB[nt] = __builtin_amdgcn_wmma_f32_16x16x32_bf16(
                               false, a1, false, b, (short)0, accB[nt], false, false);
            }
        }
    }

    // C/D layout: VGPR r -> (M = r + hi*8, N = m).
    // Fast path for full blocks (block-uniform scalar branch), guarded tail otherwise.
    if (blockRow + 256 <= N) {
        float* op = out + (rowBase + hi * 8) * COUT + m;
        #pragma unroll
        for (int mt = 0; mt < 2; ++mt) {
            #pragma unroll
            for (int nt = 0; nt < 8; ++nt) {
                const v8f acc = mt ? accB[nt] : accA[nt];
                #pragma unroll
                for (int r = 0; r < 8; ++r)
                    op[(mt * 16 + r) * COUT + nt * 16] = acc[r];
            }
        }
    } else {
        #pragma unroll
        for (int mt = 0; mt < 2; ++mt) {
            const long rb = rowBase + mt * 16 + hi * 8;
            #pragma unroll
            for (int nt = 0; nt < 8; ++nt) {
                const v8f acc = mt ? accB[nt] : accA[nt];
                #pragma unroll
                for (int r = 0; r < 8; ++r) {
                    if (rb + r < N)
                        out[(rb + r) * COUT + nt * 16 + m] = acc[r];
                }
            }
        }
    }
}

// ---------- stats stage 1: per-block column sum / sumsq ----------
__global__ __launch_bounds__(256) void stats_partial_kernel(const float* __restrict__ out,
                                                            float* __restrict__ partials,
                                                            long N, int rowsPer)
{
    __shared__ float sh[512];
    int t = threadIdx.x;
    int c = t & 127;
    int h = t >> 7;
    long r0 = (long)blockIdx.x * rowsPer;
    long r1 = r0 + rowsPer; if (r1 > N) r1 = N;
    float s = 0.f, s2 = 0.f;
    for (long r = r0 + h; r < r1; r += 2) {
        float v = out[r * COUT + c];
        s += v; s2 += v * v;
    }
    sh[t] = s; sh[256 + t] = s2;
    __syncthreads();
    if (h == 0) {
        partials[blockIdx.x * 256 + c]       = sh[c] + sh[128 + c];
        partials[blockIdx.x * 256 + 128 + c] = sh[256 + c] + sh[256 + 128 + c];
    }
}

// ---------- stats stage 2: reduce 256 partial rows -> stats[0:128]=sum, [128:256]=sumsq ----------
__global__ __launch_bounds__(256) void stats_final_kernel(const float* __restrict__ partials,
                                                          float* __restrict__ stats)
{
    int t = threadIdx.x;
    float s = 0.f;
    for (int b = 0; b < 256; ++b) s += partials[b * 256 + t];
    stats[t] = s;
}

// ---------- BN (biased var) + LeakyReLU, in place on d_out ----------
__global__ __launch_bounds__(256) void bn_apply_kernel(float* __restrict__ out,
                                                       const float* __restrict__ stats,
                                                       const float* __restrict__ gamma,
                                                       const float* __restrict__ beta,
                                                       long total, float invN)
{
    long i = ((long)blockIdx.x * blockDim.x + threadIdx.x) * 4;
    if (i >= total) return;
    int c = (int)(i & (COUT - 1));
    float4 v = *(float4*)(out + i);
    float y[4] = {v.x, v.y, v.z, v.w};
    #pragma unroll
    for (int j = 0; j < 4; ++j) {
        float mean = stats[c + j] * invN;
        float ex2  = stats[128 + c + j] * invN;
        float var  = ex2 - mean * mean;
        float rstd = rsqrtf(var + EPS);
        float z = (y[j] - mean) * rstd * gamma[c + j] + beta[c + j];
        y[j] = z > 0.f ? z : LEAK * z;
    }
    *(float4*)(out + i) = (float4){y[0], y[1], y[2], y[3]};
}

extern "C" void kernel_launch(void* const* d_in, const int* in_sizes, int n_in,
                              void* d_out, int out_size, void* d_ws, size_t ws_size,
                              hipStream_t stream)
{
    (void)n_in; (void)out_size; (void)ws_size;
    const float* feats = (const float*)d_in[0];
    const float* W     = (const float*)d_in[1];
    const float* bias  = (const float*)d_in[2];
    const float* gamma = (const float*)d_in[3];
    const float* beta  = (const float*)d_in[4];
    const int*   neigh = (const int*)d_in[5];
    float* out = (float*)d_out;

    const long N = (long)in_sizes[0] / CIN;   // 400000

    // workspace: [feats_bf16 (N+1)x128] [Wt bf16 9x128x128] [partials 256x256 f32] [stats 256 f32]
    char* ws = (char*)d_ws;
    size_t szF = (((size_t)(N + 1) * CIN * sizeof(bf16)) + 255) & ~(size_t)255;
    bf16*  fbf      = (bf16*)ws;
    bf16*  wt       = (bf16*)(ws + szF);
    float* partials = (float*)(ws + szF + (size_t)KTAPS * CIN * COUT * sizeof(bf16));
    float* stats    = partials + 256 * 256;

    const long totalF = (N + 1) * CIN;
    cvt_feats_kernel<<<(int)((totalF / 8 + 255) / 256), 256, 0, stream>>>(
        feats, fbf, N * CIN, totalF);
    cvt_w_kernel<<<(KTAPS * CIN * COUT + 255) / 256, 256, 0, stream>>>(W, wt);

    sparse_gemm_kernel<<<(int)((N + 255) / 256), 256, 0, stream>>>(
        fbf, wt, bias, neigh, out, N);

    const int rowsPer = (int)((N + 255) / 256);
    stats_partial_kernel<<<256, 256, 0, stream>>>(out, partials, N, rowsPer);
    stats_final_kernel<<<1, 256, 0, stream>>>(partials, stats);

    const long total = N * COUT;
    bn_apply_kernel<<<(int)((total / 4 + 255) / 256), 256, 0, stream>>>(
        out, stats, gamma, beta, total, 1.0f / (float)N);
}
